// MultiHeadAttention_80032420594010
// MI455X (gfx1250) — matrix operand order repaired and checked by reading, not compiled, not run on hardware
//
#include <hip/hip_runtime.h>
#include <hip/hip_bf16.h>

typedef __bf16 bf16;
typedef __attribute__((ext_vector_type(16))) __bf16 bf16x16;
typedef __attribute__((ext_vector_type(8)))  __bf16 bf16x8;
typedef __attribute__((ext_vector_type(8)))  float  f32x8;
typedef __attribute__((ext_vector_type(4)))  int    i32x4;

#define TB   2
#define TT   2048
#define TC   1024
#define TH   16
#define TD   64
#define N3C  3072

// Padded LDS row strides (elements): stride*2 bytes has gcd(stride/2,32)=4
// so the 16 fragment lanes map to disjoint bank quads; still 16B aligned.
#define PAD32 40   // for 32-wide rows
#define PAD64 72   // for 64-wide rows

// q is pre-scaled by (1/sqrt(D)) * log2(e) so attention uses exp2 directly.
#define QSCALE (0.125f * 1.44269504088896340736f)

#if __has_builtin(__builtin_amdgcn_global_load_async_to_lds_b128) && \
    __has_builtin(__builtin_amdgcn_s_wait_asynccnt)
#define USE_ASYNC_LDS 1
#endif

__device__ __forceinline__ f32x8 wmma_bf16(bf16x16 a, bf16x16 b, f32x8 c) {
  return __builtin_amdgcn_wmma_f32_16x16x32_bf16(false, a, false, b,
                                                 (short)0, c, false, false);
}

// A fragment 16x32 bf16 (row-major source, stride in elements).
// lane<16: row=lane, K {0..7},{16..23}; lane>=16: row=lane-16, K {8..15},{24..31}
__device__ __forceinline__ bf16x16 load_a_frag(const bf16* base, int stride, int lane) {
  int row = lane & 15;
  int hi  = lane >> 4;
  const bf16* p = base + row * stride + hi * 8;
  union { bf16x16 v; bf16x8 h[2]; } u;
  u.h[0] = *(const bf16x8*)(p);
  u.h[1] = *(const bf16x8*)(p + 16);
  return u.v;
}

// B fragment 32x16 bf16, source stored n-major [n][k] (stride in elements).
// lane<16: n=lane, K {0..7},{16..23}; lane>=16: n=lane-16, K {8..15},{24..31}
__device__ __forceinline__ bf16x16 load_b_frag(const bf16* base, int stride, int lane) {
  int row = lane & 15;
  int hi  = lane >> 4;
  const bf16* p = base + row * stride + hi * 8;
  union { bf16x16 v; bf16x8 h[2]; } u;
  u.h[0] = *(const bf16x8*)(p);
  u.h[1] = *(const bf16x8*)(p + 16);
  return u.v;
}

// ---------------- Kernel 1: QKV projection (bf16 WMMA GEMM) ----------------
// Block tile 64m x 128n, 4 waves, wave tile 32m x 64n (2x4 WMMA frags).
// Writes q/k/v workspace in [B,H,T,D] bf16; q scaled by QSCALE.
__global__ __launch_bounds__(128)
void qkv_gemm_kernel(const float* __restrict__ x, const float* __restrict__ w,
                     const float* __restrict__ bias,
                     bf16* __restrict__ qws, bf16* __restrict__ kws,
                     bf16* __restrict__ vws) {
  __shared__ bf16 lA[64][PAD32];    // X tile, m-major
  __shared__ bf16 lB[128][PAD32];   // W tile, n-major (transposed on load)

  const int tid   = threadIdx.x;
  const int lane  = tid & 31;
  const int wv    = tid >> 5;               // 0..3
  const int mBase = blockIdx.x * 64;        // 64 tiles over M=4096
  const int nBase = blockIdx.y * 128;       // 24 tiles over N=3072
  const int mW    = (wv & 1) * 32;
  const int nW    = (wv >> 1) * 64;

  f32x8 acc[2][4] = {};

  for (int k0 = 0; k0 < TC; k0 += 32) {
    // Stage X tile (64m x 32k): 2 units/thread, 8 contiguous k-floats each
#pragma unroll
    for (int r = 0; r < 2; ++r) {
      int u  = tid + r * 128;
      int m  = u >> 2;                 // 0..63
      int kc = (u & 3) * 8;            // 0..24
      const float* src = x + (size_t)(mBase + m) * TC + k0 + kc;
      bf16x8 v;
#pragma unroll
      for (int i = 0; i < 8; ++i) v[i] = (bf16)src[i];
      *(bf16x8*)&lA[m][kc] = v;
    }
    // Stage W tile transposed to [n][k]: 4 units/thread, 8 contiguous n-floats
#pragma unroll
    for (int r = 0; r < 4; ++r) {
      int u  = tid + r * 128;          // 0..511
      int kk = u >> 4;                 // 0..31
      int n0 = (u & 15) * 8;           // 0..120
      const float* src = w + (size_t)(k0 + kk) * N3C + nBase + n0;
#pragma unroll
      for (int i = 0; i < 8; ++i) lB[n0 + i][kk] = (bf16)src[i];
    }
    __syncthreads();

    bf16x16 a0 = load_a_frag(&lA[mW][0],      PAD32, lane);
    bf16x16 a1 = load_a_frag(&lA[mW + 16][0], PAD32, lane);
#pragma unroll
    for (int n = 0; n < 4; ++n) {
      bf16x16 bb = load_b_frag(&lB[nW + n * 16][0], PAD32, lane);
      acc[0][n] = wmma_bf16(a0, bb, acc[0][n]);
      acc[1][n] = wmma_bf16(a1, bb, acc[1][n]);
    }
    __syncthreads();
  }

  // Epilogue: bias, split q/k/v, reshape to [B,H,T,D], q pre-scaled.
  const int col = lane & 15;
  const int hi  = lane >> 4;
#pragma unroll
  for (int mi = 0; mi < 2; ++mi) {
#pragma unroll
    for (int ni = 0; ni < 4; ++ni) {
      int nGlob = nBase + nW + ni * 16 + col;
      float bv  = bias[nGlob];
      int sec   = nGlob >> 10;          // 0=q 1=k 2=v
      int cc    = nGlob & 1023;
      int h     = cc >> 6;
      int d     = cc & 63;
      bf16* dst = (sec == 0) ? qws : ((sec == 1) ? kws : vws);
      float sc  = (sec == 0) ? QSCALE : 1.0f;
#pragma unroll
      for (int r = 0; r < 8; ++r) {
        int mGlob = mBase + mW + mi * 16 + r + hi * 8;
        int b = mGlob >> 11;            // / T
        int t = mGlob & 2047;
        float v = (acc[mi][ni][r] + bv) * sc;
        dst[((size_t)(b * TH + h) * TT + t) * TD + d] = (bf16)v;
      }
    }
  }
}

// ---------------- Kernel 2: causal flash attention (bf16 WMMA) ----------------
// Block = (qTile of 64 rows) x (b,h). 4 waves, each owns 16 query rows.
__global__ __launch_bounds__(128)
void flash_attn_kernel(const bf16* __restrict__ qws, const bf16* __restrict__ kws,
                       const bf16* __restrict__ vws, float* __restrict__ out) {
  __shared__ bf16 lK[32][PAD64];       // keys x d   (row-major == B layout for Q@K^T)
  __shared__ bf16 lV[64][PAD32];       // d x keys   (n-major for P@V)
  __shared__ bf16 lP[4][16][PAD32];    // per-wave P transpose scratch

  const int tid   = threadIdx.x;
  const int lane  = tid & 31;
  const int wv    = tid >> 5;
  const int qTile = blockIdx.x;          // 0..31
  const int bh    = blockIdx.y;          // 0..31
  const int qBase = qTile * 64;
  const int col   = lane & 15;
  const int hi    = lane >> 4;

  const bf16* qhead = qws + (size_t)bh * TT * TD;
  const bf16* khead = kws + (size_t)bh * TT * TD;
  const bf16* vhead = vws + (size_t)bh * TT * TD;

  // This wave's Q fragments (16 rows x 64 d -> 2 A-frags, straight from global)
  bf16x16 aq[2];
  {
    int row = qBase + wv * 16 + col;
    const bf16* p = qhead + (size_t)row * TD;
#pragma unroll
    for (int c = 0; c < 2; ++c) {
      union { bf16x16 v; bf16x8 h[2]; } u;
      u.h[0] = *(const bf16x8*)(p + c * 32 + hi * 8);
      u.h[1] = *(const bf16x8*)(p + c * 32 + 16 + hi * 8);
      aq[c] = u.v;
    }
  }

  f32x8 o[4] = {};
  float mrow[8], lrow[8];
#pragma unroll
  for (int r = 0; r < 8; ++r) { mrow[r] = -1e30f; lrow[r] = 0.0f; }

  const int key = tid >> 2;              // 0..31 (staging role)
  const int d0  = (tid & 3) * 16;        // 0,16,32,48

  const int nTiles = qTile * 2 + 2;      // 32-key tiles covering keys <= qBase+63
  for (int kt = 0; kt < nTiles; ++kt) {
    const int kBase = kt * 32;

    // ---- Cooperative stage ----
    const bf16* ks = khead + (size_t)(kBase + key) * TD + d0;
#if USE_ASYNC_LDS
    // K tile via the CDNA5 async global->LDS engine (ASYNCcnt)
    __builtin_amdgcn_global_load_async_to_lds_b128(
        (i32x4*)(void*)ks, (i32x4*)(void*)&lK[key][d0], 0, 0);
    __builtin_amdgcn_global_load_async_to_lds_b128(
        (i32x4*)(void*)(ks + 8), (i32x4*)(void*)&lK[key][d0 + 8], 0, 0);
#else
    *(bf16x8*)&lK[key][d0]     = *(const bf16x8*)(ks);
    *(bf16x8*)&lK[key][d0 + 8] = *(const bf16x8*)(ks + 8);
#endif
    // V tile: coalesced 16B global loads, transposed scatter into LDS [d][key]
    {
      const bf16* vs = vhead + (size_t)(kBase + key) * TD + d0;
      bf16x8 v0 = *(const bf16x8*)(vs);
      bf16x8 v1 = *(const bf16x8*)(vs + 8);
#pragma unroll
      for (int i = 0; i < 8; ++i) {
        lV[d0 + i][key]     = v0[i];
        lV[d0 + 8 + i][key] = v1[i];
      }
    }
#if USE_ASYNC_LDS
    __builtin_amdgcn_s_wait_asynccnt(0);
#endif
    __syncthreads();

    // Scores: S(16x32) = Q(16x64) @ K^T, two 16-key halves
    f32x8 s[2];
#pragma unroll
    for (int nh = 0; nh < 2; ++nh) {
      f32x8 sa = {};
      sa = wmma_bf16(aq[0], load_b_frag(&lK[nh * 16][0],  PAD64, lane), sa);
      sa = wmma_bf16(aq[1], load_b_frag(&lK[nh * 16][32], PAD64, lane), sa);
      s[nh] = sa;
    }

    // Causal mask + online softmax (rows live in 16-lane halves)
    float pv0[8], pv1[8];
#pragma unroll
    for (int r = 0; r < 8; ++r) {
      int qi = qBase + wv * 16 + r + hi * 8;
      float s0 = (kBase + col      <= qi) ? s[0][r] : -1e30f;
      float s1 = (kBase + 16 + col <= qi) ? s[1][r] : -1e30f;
      float mx = fmaxf(s0, s1);
#pragma unroll
      for (int off = 1; off < 16; off <<= 1)
        mx = fmaxf(mx, __shfl_xor(mx, off, 32));
      float mn = fmaxf(mrow[r], mx);
      float alpha = exp2f(mrow[r] - mn);
      float p0 = exp2f(s0 - mn);
      float p1 = exp2f(s1 - mn);
      float rs = p0 + p1;
#pragma unroll
      for (int off = 1; off < 16; off <<= 1)
        rs += __shfl_xor(rs, off, 32);
      lrow[r] = lrow[r] * alpha + rs;
      mrow[r] = mn;
      pv0[r] = p0; pv1[r] = p1;
#pragma unroll
      for (int n = 0; n < 4; ++n) o[n][r] *= alpha;
    }

    // Transpose P into A-fragment layout via per-wave LDS scratch
#pragma unroll
    for (int r = 0; r < 8; ++r) {
      lP[wv][r + hi * 8][col]      = (bf16)pv0[r];
      lP[wv][r + hi * 8][16 + col] = (bf16)pv1[r];
    }
    __syncthreads();

    // O(16x64) += P(16x32) @ V(32x64)
    bf16x16 ap = load_a_frag(&lP[wv][0][0], PAD32, lane);
#pragma unroll
    for (int n = 0; n < 4; ++n)
      o[n] = wmma_bf16(ap, load_b_frag(&lV[n * 16][0], PAD32, lane), o[n]);
    __syncthreads();
  }

  // Normalize and write out [B,T,C] fp32, c = h*64 + d
  const int b = bh >> 4;
  const int h = bh & 15;
#pragma unroll
  for (int r = 0; r < 8; ++r) {
    int t = qBase + wv * 16 + r + hi * 8;
    float inv = 1.0f / lrow[r];
    float* dst = out + ((size_t)b * TT + t) * TC + h * TD;
#pragma unroll
    for (int n = 0; n < 4; ++n)
      dst[n * 16 + col] = o[n][r] * inv;
  }
}

extern "C" void kernel_launch(void* const* d_in, const int* in_sizes, int n_in,
                              void* d_out, int out_size, void* d_ws, size_t ws_size,
                              hipStream_t stream) {
  (void)in_sizes; (void)n_in; (void)out_size; (void)ws_size;
  const float* x    = (const float*)d_in[0];
  const float* w    = (const float*)d_in[1];
  const float* bias = (const float*)d_in[2];
  float* out = (float*)d_out;

  bf16* qws = (bf16*)d_ws;
  const size_t per = (size_t)TB * TH * TT * TD;   // 4,194,304 elems (8 MB bf16)
  bf16* kws = qws + per;
  bf16* vws = kws + per;

  // QKV projection: M=4096 (64 tiles) x N=3072 (24 tiles of 128)
  qkv_gemm_kernel<<<dim3(64, 24), 128, 0, stream>>>(x, w, bias, qws, kws, vws);
  // Flash attention: 32 query tiles x 32 (b,h) heads
  flash_attn_kernel<<<dim3(32, 32), 128, 0, stream>>>(qws, kws, vws, out);
}
